// DilatedRNNClassifier_17446157156587
// MI455X (gfx1250) — matrix operand order repaired and checked by reading, not compiled, not run on hardware
//
#include <hip/hip_runtime.h>
#include <hip/hip_bf16.h>

typedef __attribute__((ext_vector_type(16))) __bf16 v16bf;
typedef __attribute__((ext_vector_type(8)))  float  v8f;

union BfFrag { v16bf v; unsigned short s[16]; };

#define LOG2E 1.4426950408889634f

__device__ __forceinline__ unsigned short f2bf(float x) {
    unsigned int u = __float_as_uint(x);
    unsigned int r = ((u >> 16) & 1u) + 0x7FFFu;   // round-to-nearest-even
    return (unsigned short)((u + r) >> 16);
}

// Branch-free transcendentals on the v_exp_f32 / v_rcp_f32 hardware path.
__device__ __forceinline__ float fast_exp2(float x) { return __builtin_amdgcn_exp2f(x); }
__device__ __forceinline__ float fast_rcp(float x)  { return __builtin_amdgcn_rcpf(x); }
__device__ __forceinline__ float fast_exp(float x)  { return fast_exp2(x * LOG2E); }
__device__ __forceinline__ float sigmoidf(float x) {
    return fast_rcp(1.0f + fast_exp2(-LOG2E * x));     // saturates cleanly at +/-inf
}
__device__ __forceinline__ float tanh_fast(float x) {
    float s = fast_rcp(1.0f + fast_exp2(-2.0f * LOG2E * x));
    return __builtin_fmaf(2.0f, s, -1.0f);             // 2*sigmoid(2x)-1
}

// ---------------------------------------------------------------------------
// Pack an fp32 weight matrix W[N][K] (used as B[k][n] = W[n*K+k]) into bf16
// WMMA B-fragments: fragment f = kc*(N/16)+nt, laid out as 32 lanes x 16 vals.
// ---------------------------------------------------------------------------
__global__ void prep_frags(const float* __restrict__ W, int K, int N, int KT,
                           unsigned short* __restrict__ frags) {
    int i = blockIdx.x * 256 + threadIdx.x;
    int total = KT * (N / 16) * 512;
    if (i >= total) return;
    int idx  = i & 15;
    int lane = (i >> 4) & 31;
    int f    = i >> 9;
    int nt   = f % (N / 16);
    int kc   = f / (N / 16);
    int n = nt * 16 + (lane & 15);
    int k = kc * 32 + (idx < 8 ? (lane >> 4) * 8 + idx : 16 + (lane >> 4) * 8 + (idx - 8));
    float v = (k < K) ? W[(long)n * K + k] : 0.0f;
    frags[i] = f2bf(v);
}

__global__ void add_bias(const float* __restrict__ a, const float* __restrict__ b,
                         float* __restrict__ o, int n) {
    int i = blockIdx.x * 256 + threadIdx.x;
    if (i < n) o[i] = a[i] + b[i];
}

// ---------------------------------------------------------------------------
// Generic WMMA GEMM: out[M x N] = act(A @ B + bias), A rows gathered from an
// fp32 source with (rowStride, colOff, time-stride) so the same kernel does
// dilated-gather projections and the dense feed-forward GEMMs.
// One wave per 16-row M tile; A fragments assembled in registers (bf16).
// ACT: 0=none, 1=relu, 2=tanh
// ---------------------------------------------------------------------------
template <int KT, int NT, int ACT>
__global__ __launch_bounds__(256) void wmma_gemm(
    const float* __restrict__ src, int srcRowStride, int colOff, int tstride,
    int Td, int K, const unsigned short* __restrict__ Bfr,
    const float* __restrict__ bias, float* __restrict__ out, int outStride) {
    const int lane  = threadIdx.x & 31;
    const int wave  = threadIdx.x >> 5;
    const int lh    = lane >> 4;                 // lane half
    const int mTile = blockIdx.x * 8 + wave;
    const int mBase = mTile * 16;

    // Assemble A fragments (16x32 bf16 per k-chunk) in registers.
    BfFrag a[KT];
    {
        const int m = mBase + (lane & 15);
        const int b = m / Td, tt = m % Td;
        const float* row =
            src + ((long)b * 2048 + (long)tt * tstride) * srcRowStride + colOff;
#pragma unroll
        for (int kc = 0; kc < KT; ++kc) {
#pragma unroll
            for (int idx = 0; idx < 16; ++idx) {
                int k = kc * 32 + (idx < 8 ? lh * 8 + idx : 16 + lh * 8 + (idx - 8));
                float v = (k < K) ? row[k] : 0.0f;
                a[kc].s[idx] = f2bf(v);
            }
        }
    }

#pragma unroll 1
    for (int n = 0; n < NT; ++n) {
        v8f acc = {0.f, 0.f, 0.f, 0.f, 0.f, 0.f, 0.f, 0.f};
#pragma unroll
        for (int kc = 0; kc < KT; ++kc) {
            v16bf bf = *(const v16bf*)(Bfr + ((long)(kc * NT + n) * 512 + lane * 16));
            acc = __builtin_amdgcn_wmma_f32_16x16x32_bf16(
                false, a[kc].v, false, bf, (short)0, acc, false, false);
        }
#pragma unroll
        for (int r = 0; r < 8; ++r) {
            int m   = r + lh * 8;
            int col = n * 16 + (lane & 15);
            float v = acc[r] + bias[col];
            if (ACT == 1) v = fmaxf(v, 0.0f);
            else if (ACT == 2) v = tanh_fast(v);
            out[(long)(mBase + m) * outStride + col] = v;
        }
    }
}

// ---------------------------------------------------------------------------
// Persistent dilated-LSTM step kernel. Grid = 4 workgroups (16 batch rows
// each, fully independent -> no grid sync). Whh lives in VGPRs as bf16
// B-fragments (16 per wave). h lives in LDS in WMMA A-layout; c in registers.
// Per step: 16 WMMAs/wave -> gates in LDS -> elementwise -> h back to LDS.
// ---------------------------------------------------------------------------
__global__ __launch_bounds__(256) void lstm_kernel(
    const float* __restrict__ xg, const unsigned short* __restrict__ Whh,
    float* __restrict__ hout, int Td) {
    __shared__ __align__(64) unsigned short hA[4 * 512];  // 4 k-chunks, A layout
    __shared__ float gates[16 * 512];

    const int tid  = threadIdx.x;
    const int lane = tid & 31;
    const int wave = tid >> 5;
    const int lh   = lane >> 4;
    const int bTile = blockIdx.x;

    // Whh resident in registers: wave covers columns [wave*64, wave*64+64)
    v16bf Bf[4][4];
#pragma unroll
    for (int kc = 0; kc < 4; ++kc)
#pragma unroll
        for (int nn = 0; nn < 4; ++nn) {
            int f = kc * 32 + (wave * 4 + nn);
            Bf[kc][nn] = *(const v16bf*)(Whh + (long)f * 512 + lane * 16);
        }

    for (int i = tid; i < 4 * 512; i += 256) hA[i] = 0;  // h0 = 0

    const int bl = tid >> 4;   // local batch row this thread updates
    const int jb = tid & 15;   // base hidden index
    float c[8];
#pragma unroll
    for (int u = 0; u < 8; ++u) c[u] = 0.0f;

    const float* xrowBase = xg + (long)(bTile * 16 + bl) * Td * 512;
    float* houtBase = hout + (long)(bTile * 16 + bl) * Td * 128;
    __syncthreads();

    for (int t = 0; t < Td; ++t) {
        const float* xrow = xrowBase + (long)t * 512;
        if (t + 1 < Td) {  // hide next step's gate-projection row behind WMMA
            const float* xn = xrow + 512;
            __builtin_prefetch(xn + jb * 4, 0, 1);
            __builtin_prefetch(xn + 128 + jb * 4, 0, 1);
            __builtin_prefetch(xn + 256 + jb * 4, 0, 1);
            __builtin_prefetch(xn + 384 + jb * 4, 0, 1);
        }
        // ---- recurrent matmul: gates += h @ Whh^T ----
        v16bf av[4];
#pragma unroll
        for (int kc = 0; kc < 4; ++kc)
            av[kc] = *(const v16bf*)&hA[kc * 512 + lane * 16];
        v8f acc[4];
#pragma unroll
        for (int nn = 0; nn < 4; ++nn)
            acc[nn] = (v8f){0.f, 0.f, 0.f, 0.f, 0.f, 0.f, 0.f, 0.f};
#pragma unroll
        for (int kc = 0; kc < 4; ++kc)
#pragma unroll
            for (int nn = 0; nn < 4; ++nn)
                acc[nn] = __builtin_amdgcn_wmma_f32_16x16x32_bf16(
                    false, av[kc], false, Bf[kc][nn], (short)0, acc[nn], false, false);
#pragma unroll
        for (int nn = 0; nn < 4; ++nn) {
            int col = wave * 64 + nn * 16 + (lane & 15);
#pragma unroll
            for (int r = 0; r < 8; ++r)
                gates[(r + lh * 8) * 512 + col] = acc[nn][r];
        }
        __syncthreads();
        // ---- elementwise LSTM cell (branch-free v_exp/v_rcp activations) ----
#pragma unroll
        for (int u = 0; u < 8; ++u) {
            int j = jb + u * 16;
            float gi = gates[bl * 512 + j]       + xrow[j];
            float gf = gates[bl * 512 + 128 + j] + xrow[128 + j];
            float gg = gates[bl * 512 + 256 + j] + xrow[256 + j];
            float go = gates[bl * 512 + 384 + j] + xrow[384 + j];
            c[u] = sigmoidf(gf) * c[u] + sigmoidf(gi) * tanh_fast(gg);
            float h = sigmoidf(go) * tanh_fast(c[u]);
            houtBase[(long)t * 128 + j] = h;
            // write h (bf16) into WMMA A layout for the next step
            int kc = j >> 5, kr = j & 31, la, idx;
            if (kr < 8)       { la = bl;      idx = kr; }
            else if (kr < 16) { la = 16 + bl; idx = kr - 8; }
            else if (kr < 24) { la = bl;      idx = 8 + (kr - 16); }
            else              { la = 16 + bl; idx = 8 + (kr - 24); }
            hA[kc * 512 + la * 16 + idx] = f2bf(h);
        }
        __syncthreads();
    }
}

// ---------------------------------------------------------------------------
// Mask invalid steps, LayerNorm over H=128, repeat d times into comb section.
// One wave per (b, tt) row.
// ---------------------------------------------------------------------------
__global__ __launch_bounds__(256) void ln_repeat(
    const float* __restrict__ hraw, const int* __restrict__ lengths,
    const float* __restrict__ g, const float* __restrict__ bvec,
    float* __restrict__ comb, int Td, int d, int colOff) {
    int row  = blockIdx.x * 8 + (threadIdx.x >> 5);
    int lane = threadIdx.x & 31;
    int b = row / Td, tt = row % Td;
    int len = lengths[b];
    int ld = (len + d - 1) / d;
    ld = ld < 1 ? 1 : (ld > Td ? Td : ld);
    bool valid = tt < ld;
    const float* src = hraw + (long)row * 128;
    float v[4], s = 0.0f;
#pragma unroll
    for (int q = 0; q < 4; ++q) {
        v[q] = valid ? src[lane * 4 + q] : 0.0f;
        s += v[q];
    }
    for (int off = 16; off; off >>= 1) s += __shfl_xor(s, off, 32);
    float mu = s * (1.0f / 128.0f);
    float vs = 0.0f;
#pragma unroll
    for (int q = 0; q < 4; ++q) { float d0 = v[q] - mu; vs += d0 * d0; }
    for (int off = 16; off; off >>= 1) vs += __shfl_xor(vs, off, 32);
    float rs = rsqrtf(vs * (1.0f / 128.0f) + 1e-5f);
    float* dst = comb + ((long)b * 2048 + (long)tt * d) * 512 + colOff;
#pragma unroll
    for (int q = 0; q < 4; ++q) {
        int j = lane * 4 + q;
        float o = (v[q] - mu) * rs * g[j] + bvec[j];
        for (int r = 0; r < d; ++r) dst[(long)r * 512 + j] = o;
    }
}

// scores[row] = tanh-hidden[row] . Wa2 + ba2  (one wave per row)
__global__ __launch_bounds__(256) void scores_kernel(
    const float* __restrict__ hidden, const float* __restrict__ Wa2,
    const float* __restrict__ ba2, float* __restrict__ scores, int M) {
    int row  = blockIdx.x * 8 + (threadIdx.x >> 5);
    int lane = threadIdx.x & 31;
    if (row >= M) return;
    const float* src = hidden + (long)row * 128;
    float s = 0.0f;
#pragma unroll
    for (int q = 0; q < 4; ++q) { int j = lane * 4 + q; s += src[j] * Wa2[j]; }
    for (int off = 16; off; off >>= 1) s += __shfl_xor(s, off, 32);
    if (lane == 0) scores[row] = s + ba2[0];
}

// masked softmax over T, attention pooling, final linear. One block per batch.
__global__ __launch_bounds__(128) void pool_kernel(
    const float* __restrict__ fused, const float* __restrict__ scores,
    const int* __restrict__ lengths, const float* __restrict__ Wc,
    const float* __restrict__ bc, float* __restrict__ out) {
    int b = blockIdx.x, tid = threadIdx.x;
    int len = lengths[b];
    __shared__ float red[128];
    float mx = -1e30f;
    for (int t = tid; t < 2048; t += 128) {
        float s = (t < len) ? scores[b * 2048 + t] : -1e9f;
        mx = fmaxf(mx, s);
    }
    red[tid] = mx; __syncthreads();
    for (int o = 64; o; o >>= 1) { if (tid < o) red[tid] = fmaxf(red[tid], red[tid + o]); __syncthreads(); }
    mx = red[0]; __syncthreads();
    float sm = 0.0f;
    for (int t = tid; t < 2048; t += 128) {
        float s = (t < len) ? scores[b * 2048 + t] : -1e9f;
        sm += fast_exp(s - mx);
    }
    red[tid] = sm; __syncthreads();
    for (int o = 64; o; o >>= 1) { if (tid < o) red[tid] += red[tid + o]; __syncthreads(); }
    float inv = fast_rcp(red[0]); __syncthreads();
    float acc = 0.0f;
    for (int t = 0; t < 2048; ++t) {
        float s = (t < len) ? scores[b * 2048 + t] : -1e9f;
        acc += fast_exp(s - mx) * inv * fused[((long)b * 2048 + t) * 128 + tid];
    }
    red[tid] = acc * Wc[tid]; __syncthreads();
    for (int o = 64; o; o >>= 1) { if (tid < o) red[tid] += red[tid + o]; __syncthreads(); }
    if (tid == 0) out[b] = red[0] + bc[0];
}

// ---------------------------------------------------------------------------
extern "C" void kernel_launch(void* const* d_in, const int* in_sizes, int n_in,
                              void* d_out, int out_size, void* d_ws, size_t ws_size,
                              hipStream_t stream) {
    (void)in_sizes; (void)n_in; (void)out_size; (void)ws_size;
    const float* x       = (const float*)d_in[0];
    const int*   lengths = (const int*)d_in[1];
    const float *Wih[4], *Whh[4], *bih[4], *bhh[4], *lng[4], *lnb[4];
    for (int i = 0; i < 4; ++i) {
        Wih[i] = (const float*)d_in[2 + 6 * i];
        Whh[i] = (const float*)d_in[3 + 6 * i];
        bih[i] = (const float*)d_in[4 + 6 * i];
        bhh[i] = (const float*)d_in[5 + 6 * i];
        lng[i] = (const float*)d_in[6 + 6 * i];
        lnb[i] = (const float*)d_in[7 + 6 * i];
    }
    const float* Wf1 = (const float*)d_in[26]; const float* bf1 = (const float*)d_in[27];
    const float* Wf2 = (const float*)d_in[28]; const float* bf2 = (const float*)d_in[29];
    const float* Wa1 = (const float*)d_in[30]; const float* ba1 = (const float*)d_in[31];
    const float* Wa2 = (const float*)d_in[32]; const float* ba2 = (const float*)d_in[33];
    const float* Wc  = (const float*)d_in[34]; const float* bc  = (const float*)d_in[35];

    char* ws = (char*)d_ws;
    size_t off = 0;
    auto alloc = [&](size_t bytes) -> char* {
        char* p = ws + off;
        off = (off + bytes + 255) & ~(size_t)255;
        return p;
    };
    unsigned short* fWih[4];
    fWih[0] = (unsigned short*)alloc((size_t)1 * 32 * 512 * 2);
    for (int i = 1; i < 4; ++i) fWih[i] = (unsigned short*)alloc((size_t)4 * 32 * 512 * 2);
    unsigned short* fWhh[4];
    for (int i = 0; i < 4; ++i) fWhh[i] = (unsigned short*)alloc((size_t)4 * 32 * 512 * 2);
    unsigned short* fWf1 = (unsigned short*)alloc((size_t)16 * 8 * 512 * 2);
    unsigned short* fWf2 = (unsigned short*)alloc((size_t)4 * 8 * 512 * 2);
    unsigned short* fWa1 = (unsigned short*)alloc((size_t)4 * 8 * 512 * 2);
    float* biasSum[4];
    for (int i = 0; i < 4; ++i) biasSum[i] = (float*)alloc(512 * 4);
    float* xg   = (float*)alloc((size_t)64 * 2048 * 512 * 4);  // 256 MB, reused below
    float* hraw = (float*)alloc((size_t)64 * 2048 * 128 * 4);  // 64 MB
    float* comb = (float*)alloc((size_t)64 * 2048 * 512 * 4);  // 256 MB
    // stage-2 aliases into the (now free) xg region:
    float* ftmp   = xg;
    float* fusedB = xg + (size_t)64 * 2048 * 128;
    float* hidden = xg + (size_t)64 * 2048 * 256;
    float* scores = xg + (size_t)64 * 2048 * 384;

    // weight packing
    prep_frags<<<64, 256, 0, stream>>>(Wih[0], 16, 512, 1, fWih[0]);
    for (int i = 1; i < 4; ++i)
        prep_frags<<<256, 256, 0, stream>>>(Wih[i], 128, 512, 4, fWih[i]);
    for (int i = 0; i < 4; ++i)
        prep_frags<<<256, 256, 0, stream>>>(Whh[i], 128, 512, 4, fWhh[i]);
    prep_frags<<<256, 256, 0, stream>>>(Wf1, 512, 128, 16, fWf1);
    prep_frags<<<64, 256, 0, stream>>>(Wf2, 128, 128, 4, fWf2);
    prep_frags<<<64, 256, 0, stream>>>(Wa1, 128, 128, 4, fWa1);
    for (int i = 0; i < 4; ++i)
        add_bias<<<2, 256, 0, stream>>>(bih[i], bhh[i], biasSum[i], 512);

    const int DIL[4] = {1, 2, 4, 8};
    for (int i = 0; i < 4; ++i) {
        int d = DIL[i], Td = 2048 / d;
        if (i == 0)
            wmma_gemm<1, 32, 0><<<Td / 2, 256, 0, stream>>>(
                x, 16, 0, 1, Td, 16, fWih[0], biasSum[0], xg, 512);
        else
            wmma_gemm<4, 32, 0><<<Td / 2, 256, 0, stream>>>(
                comb, 512, 128 * (i - 1), d, Td, 128, fWih[i], biasSum[i], xg, 512);
        lstm_kernel<<<4, 256, 0, stream>>>(xg, fWhh[i], hraw, Td);
        ln_repeat<<<(64 * Td) / 8, 256, 0, stream>>>(
            hraw, lengths, lng[i], lnb[i], comb, Td, d, 128 * i);
    }
    wmma_gemm<16, 8, 1><<<1024, 256, 0, stream>>>(comb, 512, 0, 1, 2048, 512, fWf1, bf1, ftmp, 128);
    wmma_gemm<4, 8, 0><<<1024, 256, 0, stream>>>(ftmp, 128, 0, 1, 2048, 128, fWf2, bf2, fusedB, 128);
    wmma_gemm<4, 8, 2><<<1024, 256, 0, stream>>>(fusedB, 128, 0, 1, 2048, 128, fWa1, ba1, hidden, 128);
    scores_kernel<<<131072 / 8, 256, 0, stream>>>(hidden, Wa2, ba2, scores, 131072);
    pool_kernel<<<64, 128, 0, stream>>>(fusedB, scores, lengths, Wc, bc, (float*)d_out);
}
